// LSTM_AD_33655363732368
// MI455X (gfx1250) — matrix operand order repaired and checked by reading, not compiled, not run on hardware
//
#include <hip/hip_runtime.h>
#include <hip/hip_bf16.h>

// ---------------------------------------------------------------------------
// Problem constants (from reference): T=50000, F=128, H=64, D=6F=768
// Only lstm2 + fc affect the output; lstm1 is dead code and is skipped.
// ---------------------------------------------------------------------------
#define T_STEPS 50000
#define DIN     768      // D = 6*F
#define FF      128      // F
#define G2      512      // 4*F gates of lstm2

typedef __attribute__((ext_vector_type(16))) __bf16 v16bf;
typedef __attribute__((ext_vector_type(8)))  float  v8f;

// A operand K order used here (contiguous, not AMD's table order): elements 0..15 = K sequential,
// K base = (lane<16 ? 0 : 16).
__device__ __forceinline__ v16bf pack16_a(float4 a0, float4 a1, float4 a2, float4 a3) {
    v16bf v;
    v[0]  = (__bf16)a0.x; v[1]  = (__bf16)a0.y; v[2]  = (__bf16)a0.z; v[3]  = (__bf16)a0.w;
    v[4]  = (__bf16)a1.x; v[5]  = (__bf16)a1.y; v[6]  = (__bf16)a1.z; v[7]  = (__bf16)a1.w;
    v[8]  = (__bf16)a2.x; v[9]  = (__bf16)a2.y; v[10] = (__bf16)a2.z; v[11] = (__bf16)a2.w;
    v[12] = (__bf16)a3.x; v[13] = (__bf16)a3.y; v[14] = (__bf16)a3.z; v[15] = (__bf16)a3.w;
    return v;
}

// B operand K order used here (contiguous, same as A): elements 0..15 = K sequential,
// K base = (lane<16 ? 0 : 16). Four contiguous float4s.
__device__ __forceinline__ v16bf pack16_b(float4 b0, float4 b1, float4 b2, float4 b3) {
    v16bf v;
    v[0]  = (__bf16)b0.x; v[1]  = (__bf16)b0.y; v[2]  = (__bf16)b0.z; v[3]  = (__bf16)b0.w;
    v[4]  = (__bf16)b1.x; v[5]  = (__bf16)b1.y; v[6]  = (__bf16)b1.z; v[7]  = (__bf16)b1.w;
    v[8]  = (__bf16)b2.x; v[9]  = (__bf16)b2.y; v[10] = (__bf16)b2.z; v[11] = (__bf16)b2.w;
    v[12] = (__bf16)b3.x; v[13] = (__bf16)b3.y; v[14] = (__bf16)b3.z; v[15] = (__bf16)b3.w;
    return v;
}

// Fast activations: v_exp_f32 + v_rcp_f32, no IEEE-div expansion.
__device__ __forceinline__ float sigmoid_f(float x) {
    return __builtin_amdgcn_rcpf(1.0f + __expf(-x));
}
__device__ __forceinline__ float tanh_f(float x) {
    // (e-1)/(e+1) = 1 - 2/(e+1),  e = exp(2x)
    return 1.0f - 2.0f * __builtin_amdgcn_rcpf(__expf(2.0f * x) + 1.0f);
}

// ---------------------------------------------------------------------------
// WMMA GEMM:  Out(T x N) = act( X(T x DK) @ W(N x DK)^T + B1 [+ B2] )
// One wave per 16x16 output tile; bf16 inputs, f32 accumulation.
// blockDim = 256 (8 waves); gridDim.x = T/16 m-tiles; gridDim.y * 8 n-tiles.
// ---------------------------------------------------------------------------
template <int DK, int N, int SIG, bool HASB2>
__global__ __launch_bounds__(256)
void wmma_gemm_bias(const float* __restrict__ X, const float* __restrict__ W,
                    const float* __restrict__ B1, const float* __restrict__ B2,
                    float* __restrict__ Out) {
    const int wave = threadIdx.x >> 5;
    const int lane = threadIdx.x & 31;
    const int lh   = lane & 15;     // row (A) / col (B) within tile
    const int hi   = lane >> 4;     // lane half selects K sub-range
    const int n0   = (blockIdx.y * 8 + wave) * 16;
    const int m0   = blockIdx.x * 16;

    const float* xr = X + (size_t)(m0 + lh) * DK;
    const float* wr = W + (size_t)(n0 + lh) * DK;

    // Every element this lane holds in C has column N = n0 + lh.
    float bb = B1[n0 + lh];
    if (HASB2) bb += B2[n0 + lh];
    v8f acc;
#pragma unroll
    for (int r = 0; r < 8; ++r) acc[r] = bb;

    const int akoff = hi ? 16 : 0;   // A: contiguous K half
    const int bkoff = hi ? 16 : 0;   // B: contiguous K half
#pragma unroll 4
    for (int k0 = 0; k0 < DK; k0 += 32) {
        const float* xa = xr + k0 + akoff;
        const float* wb = wr + k0 + bkoff;
        float4 ax0 = *(const float4*)(xa + 0);
        float4 ax1 = *(const float4*)(xa + 4);
        float4 ax2 = *(const float4*)(xa + 8);
        float4 ax3 = *(const float4*)(xa + 12);
        float4 bx0 = *(const float4*)(wb + 0);
        float4 bx1 = *(const float4*)(wb + 4);
        float4 bx2 = *(const float4*)(wb + 8);
        float4 bx3 = *(const float4*)(wb + 12);
        v16bf a = pack16_a(ax0, ax1, ax2, ax3);
        v16bf b = pack16_b(bx0, bx1, bx2, bx3);
        acc = __builtin_amdgcn_wmma_f32_16x16x32_bf16(
                  false, a, false, b, (short)0, acc, false, false);
    }

    // C layout: VGPR r -> M = m0 + r + hi*8, N = n0 + lh
    float* obase = Out + (size_t)(m0 + hi * 8) * N + (n0 + lh);
#pragma unroll
    for (int r = 0; r < 8; ++r) {
        float v = acc[r];
        if (SIG) v = sigmoid_f(v);
        obase[(size_t)r * N] = v;
    }
}

// ---------------------------------------------------------------------------
// LSTM-2 recurrence: single workgroup (512 threads = 16 wave32s) on one WGP.
// Whh2 (512x128 f32) lives in LDS (266 KB of the 320 KB WGP pool), padded to
// stride 132 floats so per-lane float4 reads of consecutive rows spread banks.
// Thread t computes gate row t (128-MAC dot with h); threads 0..127 own c_j.
// pre2 rows are prefetched 4 steps ahead to hide HBM latency.
// ---------------------------------------------------------------------------
#define WPAD 132
#define SCAN_SMEM_FLOATS (512 * WPAD + 128 + 512)

__global__ __launch_bounds__(512)
void lstm2_scan(const float* __restrict__ pre2, const float* __restrict__ Whh2,
                const float* __restrict__ h0, const float* __restrict__ c0,
                float* __restrict__ hs2) {
    extern __shared__ float smem[];
    float* wlds = smem;                 // 512 * 132
    float* hlds = smem + 512 * WPAD;    // 128
    float* glds = hlds + 128;           // 512

    const int t = threadIdx.x;

    // Stage Whh2 into LDS, global-coalesced.
    for (int i = t; i < G2 * FF; i += 512) {
        int r = i >> 7, k = i & 127;
        wlds[r * WPAD + k] = Whh2[i];
    }
    if (t < FF) hlds[t] = h0[t];
    float c = (t < FF) ? c0[t] : 0.0f;
    __syncthreads();

    const float4* wrow = (const float4*)(wlds + t * WPAD);
    const float4* h4   = (const float4*)hlds;
    const bool isG = (t >= 256) && (t < 384);

    auto ld = [&](int s) -> float {
        return (s < T_STEPS) ? pre2[(size_t)s * G2 + t] : 0.0f;
    };

    auto body = [&](int s, float pre) {
        float acc = pre;
#pragma unroll
        for (int kk = 0; kk < FF / 4; ++kk) {
            float4 w = wrow[kk];
            float4 h = h4[kk];          // broadcast LDS read
            acc += w.x * h.x + w.y * h.y + w.z * h.z + w.w * h.w;
        }
        float g = isG ? tanh_f(acc) : sigmoid_f(acc);
        glds[t] = g;
        __syncthreads();
        if (t < FF) {
            float iv = glds[t], fv = glds[t + 128], gv = glds[t + 256], ov = glds[t + 384];
            c = fv * c + iv * gv;
            float hv = ov * tanh_f(c);
            hlds[t] = hv;
            hs2[(size_t)s * FF + t] = hv;   // fire-and-forget store
        }
        __syncthreads();
    };

    float pf0 = ld(0), pf1 = ld(1), pf2 = ld(2), pf3 = ld(3);
    for (int step = 0; step < T_STEPS; step += 4) {
        body(step + 0, pf0); pf0 = ld(step + 4);
        body(step + 1, pf1); pf1 = ld(step + 5);
        body(step + 2, pf2); pf2 = ld(step + 6);
        body(step + 3, pf3); pf3 = ld(step + 7);
    }
}

// ---------------------------------------------------------------------------
// Launch: pre2 GEMM (WMMA) -> scan (one WGP, big LDS) -> fc GEMM+sigmoid (WMMA)
// Inputs (setup_inputs order): 0:x 1:h1 2:c1 3:h2 4:c2 5:Wih1 6:Whh1 7:bih1
// 8:bhh1 9:Wih2 10:Whh2 11:bih2 12:bhh2 13:Wfc 14:bfc   (all float32)
// ---------------------------------------------------------------------------
extern "C" void kernel_launch(void* const* d_in, const int* in_sizes, int n_in,
                              void* d_out, int out_size, void* d_ws, size_t ws_size,
                              hipStream_t stream) {
    (void)in_sizes; (void)n_in; (void)out_size; (void)ws_size;

    const float* x    = (const float*)d_in[0];
    const float* h2   = (const float*)d_in[3];
    const float* c2   = (const float*)d_in[4];
    const float* Wih2 = (const float*)d_in[9];
    const float* Whh2 = (const float*)d_in[10];
    const float* bih2 = (const float*)d_in[11];
    const float* bhh2 = (const float*)d_in[12];
    const float* Wfc  = (const float*)d_in[13];
    const float* bfc  = (const float*)d_in[14];

    float* pre2 = (float*)d_ws;                                   // T x 512 f32
    float* hs2  = (float*)((char*)d_ws + (size_t)T_STEPS * G2 * sizeof(float)); // T x 128

    // 1) pre2 = x @ Wih2^T + (bih2 + bhh2)     [WMMA bf16, f32 accum]
    wmma_gemm_bias<DIN, G2, 0, true><<<dim3(T_STEPS / 16, 4), dim3(256), 0, stream>>>(
        x, Wih2, bih2, bhh2, pre2);

    // 2) sequential LSTM recurrence, Whh2 resident in LDS
    size_t smem = (size_t)SCAN_SMEM_FLOATS * sizeof(float);
    hipFuncSetAttribute((const void*)lstm2_scan,
                        hipFuncAttributeMaxDynamicSharedMemorySize, (int)smem);
    lstm2_scan<<<dim3(1), dim3(512), smem, stream>>>(pre2, Whh2, h2, c2, hs2);

    // 3) out = sigmoid(hs2 @ Wfc^T + bfc)      [WMMA bf16, fused sigmoid]
    wmma_gemm_bias<FF, FF, 1, false><<<dim3(T_STEPS / 16, 1), dim3(256), 0, stream>>>(
        hs2, Wfc, bfc, nullptr, (float*)d_out);
}
